// Classifier_48266842472794
// MI455X (gfx1250) — matrix operand order repaired and checked by reading, not compiled, not run on hardware
//
#include <hip/hip_runtime.h>

// ---------------- problem sizes ----------------
#define N_SAMP 65536
#define DIM    768
#define HID    384
#define NEXP   8
#define NCLS   8
#define MT     64          // gathered rows per block
#define KB     64          // K block (2 WMMA k-steps of 32)
#define AST    72          // LDS stride (f16) for A and Wt (64+8 pad)
#define HST    392         // LDS stride (f16) for H and W2t (384+8 pad)

typedef _Float16 h8  __attribute__((ext_vector_type(8)));
typedef _Float16 h16 __attribute__((ext_vector_type(16)));
typedef float    f8  __attribute__((ext_vector_type(8)));

__device__ __forceinline__ h16 cat16(h8 lo, h8 hi) {
  return __builtin_shufflevector(lo, hi, 0,1,2,3,4,5,6,7,8,9,10,11,12,13,14,15);
}

// ---------------- routing kernels ----------------
// ws int layout: [0..7]=counts [8..15]=offsets [16..23]=cursor [32..32+N)=perm
__global__ void moe_zero_k(int* ws) {
  if (threadIdx.x < 2 * NEXP) ws[threadIdx.x < NEXP ? threadIdx.x : (8 + threadIdx.x)] = 0;
}

__global__ void moe_count_k(const int* __restrict__ qt, int* __restrict__ ws) {
  int i = blockIdx.x * blockDim.x + threadIdx.x;
  if (i < N_SAMP) atomicAdd(&ws[qt[i]], 1);
}

__global__ void moe_scan_k(int* ws) {
  if (threadIdx.x == 0) {
    int acc = 0;
    for (int e = 0; e < NEXP; ++e) { ws[8 + e] = acc; ws[16 + e] = acc; acc += ws[e]; }
  }
}

__global__ void moe_scatter_k(const int* __restrict__ qt, int* __restrict__ ws,
                              int* __restrict__ perm) {
  int i = blockIdx.x * blockDim.x + threadIdx.x;
  if (i < N_SAMP) { int p = atomicAdd(&ws[16 + qt[i]], 1); perm[p] = i; }
}

// W1 f32 -> f16 (same [E][D][H] layout), 4 elems/thread
__global__ void moe_cvt_w1_k(const float* __restrict__ W1, _Float16* __restrict__ W1h) {
  int i = (blockIdx.x * blockDim.x + threadIdx.x) * 4;
  if (i < NEXP * DIM * HID) {
    #pragma unroll
    for (int j = 0; j < 4; ++j) W1h[i + j] = (_Float16)W1[i + j];
  }
}

// ---------------- fused 2-layer WMMA GEMM ----------------
// 384 threads = 12 wave32. Layer1: waves tile 64x384 as (wm=wid&3 -> 16 rows,
// wn=wid>>2 -> 128 cols = 8 n-tiles). Layer2: waves 0..3 do 64x16.
__global__ __launch_bounds__(384, 1)
void moe_gemm_k(const float* __restrict__ x,  const float* __restrict__ W1,
                const _Float16* __restrict__ W1h, const int useF16,
                const float* __restrict__ b1, const float* __restrict__ W2,
                const float* __restrict__ b2, const int* __restrict__ perm,
                const int* __restrict__ counts, const int* __restrict__ offs,
                float* __restrict__ out)
{
  const int e = blockIdx.x;
  const int t = blockIdx.y;
  const int cnt = counts[e];
  if (t * MT >= cnt) return;                 // block-uniform early exit
  const int base = offs[e];

  __shared__ __align__(16) char smem[64512];
  _Float16* As  = (_Float16*)smem;               // [64][AST]   x tile (f16)
  _Float16* Ws  = (_Float16*)(smem + 9216);      // [384][AST]  W1 K-block, TRANSPOSED: Ws[h][k]
  _Float16* Hs  = (_Float16*)smem;               // [64][HST]   relu(h) (overlaps As/Ws)
  _Float16* W2s = (_Float16*)(smem + 50176);     // [16][HST]   W2 transposed+padded: W2s[c][k]

  const int tid   = threadIdx.x;
  const int lane  = tid & 31;
  const int wid   = tid >> 5;       // 0..11
  const int wm    = wid & 3;        // M group (16 rows)
  const int wn    = wid >> 2;       // N group (128 cols)
  const int mbase = wm * 16;
  const int lhalf = lane & 15;
  const int lhi   = lane >> 4;

  f8 acc[8];
  #pragma unroll
  for (int tl = 0; tl < 8; ++tl)
    #pragma unroll
    for (int i = 0; i < 8; ++i) acc[tl][i] = 0.f;

  // A-loader assignment (threads 0..255): 4 threads per row, 16 f32 each
  const int aRow = tid >> 2, aQ = tid & 3;
  int p  = t * MT + aRow;
  int pc = p < cnt ? p : cnt - 1;                    // clamp tail rows (loads only)
  const float* xrow = x + (size_t)perm[base + pc] * DIM;

  const float*    w1e  = W1  + (size_t)e * DIM * HID;
  const _Float16* w1eh = W1h + (size_t)e * DIM * HID;

  for (int kb = 0; kb < DIM / KB; ++kb) {
    const int k0 = kb * KB;
    __syncthreads();
    // ---- stage A tile (f32 -> f16) ----
    if (tid < 256) {
      const float4* src = (const float4*)(xrow + k0 + aQ * 16);
      _Float16* dst = As + aRow * AST + aQ * 16;
      #pragma unroll
      for (int v = 0; v < 4; ++v) {
        float4 f = src[v];
        dst[v*4+0] = (_Float16)f.x; dst[v*4+1] = (_Float16)f.y;
        dst[v*4+2] = (_Float16)f.z; dst[v*4+3] = (_Float16)f.w;
      }
    }
    // ---- stage W1 K-block transposed: Ws[h][kk] ----
    {
      const int h = tid;                              // 0..383 == HID
      #pragma unroll
      for (int k8 = 0; k8 < 8; ++k8) {
        h8 tmp;
        if (useF16) {
          #pragma unroll
          for (int j = 0; j < 8; ++j)
            tmp[j] = w1eh[(size_t)(k0 + k8*8 + j) * HID + h];
        } else {
          #pragma unroll
          for (int j = 0; j < 8; ++j)
            tmp[j] = (_Float16)w1e[(size_t)(k0 + k8*8 + j) * HID + h];
        }
        *(h8*)(Ws + h * AST + k8 * 8) = tmp;
      }
    }
    __syncthreads();

    // ---- prefetch next K-block toward WGP$/L2 under the WMMA burst ----
    // next W K-block is one contiguous region: 384 threads x 1 cacheline covers it
    if (kb + 1 < DIM / KB) {
      const int kn = k0 + KB;
      if (useF16) __builtin_prefetch((const char*)w1eh + (size_t)kn * HID * 2 + tid * 128, 0, 3);
      else        __builtin_prefetch((const char*)w1e  + (size_t)kn * HID * 4 + tid * 256, 0, 3);
      if (tid < 256) __builtin_prefetch(xrow + kn + aQ * 16, 0, 3);
    }

    // ---- compute: per k-step, load ALL 8 B-frags (16x ds_load_b128) into
    //      live registers, then burst 8 back-to-back WMMAs. 64 live B-VGPRs
    //      prevent the RA from collapsing to a serialized load->wait->wmma. ----
    #pragma unroll
    for (int ks = 0; ks < 2; ++ks) {
      const int kk = ks * 32;
      // A frag: lanes 0-15 rows, K = kg*8..+7 and +16..+23
      const _Float16* arow = As + (mbase + lhalf) * AST + kk + lhi * 8;
      h16 a = cat16(*(const h8*)arow, *(const h8*)(arow + 16));
      h16 bfr[8];
      #pragma unroll
      for (int tl = 0; tl < 8; ++tl) {
        // B frag: per-lane column, K = kg*8..+7 and +16..+23 (Ws is transposed)
        const _Float16* brow = Ws + (wn*128 + tl*16 + lhalf) * AST + kk + lhi * 8;
        bfr[tl] = cat16(*(const h8*)brow, *(const h8*)(brow + 16));
      }
      #pragma unroll
      for (int tl = 0; tl < 8; ++tl)
        acc[tl] = __builtin_amdgcn_wmma_f32_16x16x32_f16(
            false, a, false, bfr[tl], (short)0, acc[tl], false, false);
    }
  }
  __syncthreads();

  // ---- bias + ReLU -> Hs (f16), and stage W2 transposed+padded ----
  #pragma unroll
  for (int tl = 0; tl < 8; ++tl) {
    int col = wn * 128 + tl * 16 + lhalf;
    float bv = b1[e * HID + col];
    #pragma unroll
    for (int i = 0; i < 8; ++i) {
      int row = mbase + lhi * 8 + i;           // C layout: vgpr i, halves of wave
      float v = acc[tl][i] + bv;
      Hs[row * HST + col] = (_Float16)(v > 0.f ? v : 0.f);
    }
  }
  {
    const int k = tid;                          // 0..383
    const float* w2row = W2 + ((size_t)e * HID + k) * NCLS;
    #pragma unroll
    for (int c = 0; c < NCLS; ++c) W2s[c * HST + k] = (_Float16)w2row[c];
    #pragma unroll
    for (int c = NCLS; c < 16; ++c) W2s[c * HST + k] = (_Float16)0.f;
  }
  __syncthreads();

  // ---- layer 2: [64x384] @ [384x16pad], waves 0..3 (wave-uniform branch) ----
  if (wid < 4) {
    f8 acc2;
    #pragma unroll
    for (int i = 0; i < 8; ++i) acc2[i] = 0.f;
    const _Float16* ab = Hs  + (mbase + lhalf) * HST + lhi * 8;
    const _Float16* bb = W2s + lhalf * HST + lhi * 8;
    #pragma unroll
    for (int k2 = 0; k2 < HID / 32; ++k2) {
      h16 a = cat16(*(const h8*)(ab + k2 * 32), *(const h8*)(ab + k2 * 32 + 16));
      h16 b = cat16(*(const h8*)(bb + k2 * 32), *(const h8*)(bb + k2 * 32 + 16));
      acc2 = __builtin_amdgcn_wmma_f32_16x16x32_f16(
          false, a, false, b, (short)0, acc2, false, false);
    }
    const int n = lhalf;
    if (n < NCLS) {                             // after all WMMAs; divergence OK
      float bv = b2[e * NCLS + n];
      #pragma unroll
      for (int i = 0; i < 8; ++i) {
        int r  = mbase + lhi * 8 + i;
        int pp = t * MT + r;
        if (pp < cnt) {
          int orig = perm[base + pp];
          out[(size_t)orig * NCLS + n] = acc2[i] + bv;
        }
      }
    }
  }
}

// ---------------- host launcher ----------------
extern "C" void kernel_launch(void* const* d_in, const int* in_sizes, int n_in,
                              void* d_out, int out_size, void* d_ws, size_t ws_size,
                              hipStream_t stream) {
  const float* x  = (const float*)d_in[0];
  const float* W1 = (const float*)d_in[1];
  const float* b1 = (const float*)d_in[2];
  const float* W2 = (const float*)d_in[3];
  const float* b2 = (const float*)d_in[4];
  const int*   qt = (const int*)  d_in[5];
  float* out = (float*)d_out;

  int* ws   = (int*)d_ws;
  int* perm = ws + 32;
  const size_t permBytes = 128 + (size_t)N_SAMP * 4;            // 262272
  _Float16* W1h = (_Float16*)((char*)d_ws + permBytes);
  const size_t w1hBytes = (size_t)NEXP * DIM * HID * 2;         // ~4.7 MB
  const int useF16 = (ws_size >= permBytes + w1hBytes) ? 1 : 0; // fixed per run

  moe_zero_k<<<1, 32, 0, stream>>>(ws);
  moe_count_k<<<N_SAMP / 256, 256, 0, stream>>>(qt, ws);
  moe_scan_k<<<1, 1, 0, stream>>>(ws);
  moe_scatter_k<<<N_SAMP / 256, 256, 0, stream>>>(qt, ws, perm);
  if (useF16) {
    int total = NEXP * DIM * HID / 4;
    moe_cvt_w1_k<<<(total + 255) / 256, 256, 0, stream>>>(W1, W1h);
  }
  moe_gemm_k<<<dim3(NEXP, N_SAMP / MT), 384, 0, stream>>>(
      x, W1, W1h, useF16, b1, W2, b2, perm, ws, ws + 8, out);
}
